// GANN_86406152061412
// MI455X (gfx1250) — matrix operand order repaired and checked by reading, not compiled, not run on hardware
//
#include <hip/hip_runtime.h>
#include <hip/hip_bf16.h>

#define N_NODES  50000
#define N_EDGES  200000
#define N_GRAPHS 1024
#define HEADS    8
#define EPSBN    1e-5f
#define NEG_SLOPE 0.2f

typedef __attribute__((ext_vector_type(16))) _Float16 v16h;
typedef __attribute__((ext_vector_type(8)))  _Float16 v8h;
typedef __attribute__((ext_vector_type(8)))  float    v8f;

__device__ __forceinline__ void atomicMaxF(float* addr, float v) {
    // monotonic int/uint mapping trick; works for mixed signs and -inf init
    if (v >= 0.f) atomicMax((int*)addr, __float_as_int(v));
    else          atomicMin((unsigned int*)addr, __float_as_uint(v));
}

__device__ __forceinline__ float leakyf(float x) { return x > 0.f ? x : NEG_SLOPE * x; }

__device__ __forceinline__ v16h cat8(v8h lo, v8h hi) {
  return __builtin_shufflevector(lo, hi, 0,1,2,3,4,5,6,7,8,9,10,11,12,13,14,15);
}

// ---------------------------------------------------------------------------
// WMMA GEMM, N-register-blocked, K fully unrolled (K is a template param so
// every k-offset folds into the global_load_b128 instruction offset and the
// whole body is straight-line SSA: no address arithmetic, no buffer-rotation
// copies, loads hoisted across WMMAs by the scheduler).
// C[n,M] = A[n,K] * Bt[M,K]^T ; one wave computes a 16 x (16*NB) strip.
// Fragment layouts per CDNA5 ISA 7.12.2:
//  A lane L<16: row m0+L, K = {k0..k0+7, k0+16..k0+23}; L>=16: +8 on both K groups
//  B lane L<16: col n0+L, K = {k0..k0+7, k0+16..k0+23}; L>=16: +8 on both K groups
//  C vgpr v:    row m0+v (+8 for lanes 16..31), col n0+(L&15)
// ---------------------------------------------------------------------------
template <int NB, int K>
__global__ void __launch_bounds__(256) wmma_gemm_f16(
    const _Float16* __restrict__ A, const _Float16* __restrict__ Bt,
    float* __restrict__ C, int nRows, int M) {
  const int lane = threadIdx.x & 31;
  const int wave = threadIdx.x >> 5;
  const int nCG = M / (16 * NB);
  const int nGroups = (nRows >> 4) * nCG;
  const int grp = blockIdx.x * 8 + wave;
  if (grp >= nGroups) return;                 // wave-uniform: EXEC stays all-1s
  const int rt = grp / nCG, cg = grp % nCG;
  const int m0 = rt << 4, n0 = cg * (16 * NB);
  const int hi = (lane & 16);
  const _Float16* arow = A + (size_t)(m0 + (lane & 15)) * K + (hi ? 8 : 0);
  const _Float16* brow[NB];
  #pragma unroll
  for (int j = 0; j < NB; ++j)
    brow[j] = Bt + (size_t)(n0 + j * 16 + (lane & 15)) * K + (hi ? 8 : 0);

  v8f acc[NB];
  #pragma unroll
  for (int j = 0; j < NB; ++j) acc[j] = (v8f){};

  #pragma unroll
  for (int k0 = 0; k0 < K; k0 += 32) {
    v16h af = cat8(*(const v8h*)(arow + k0), *(const v8h*)(arow + k0 + 16));
    #pragma unroll
    for (int j = 0; j < NB; ++j) {
      v16h bf = cat8(*(const v8h*)(brow[j] + k0), *(const v8h*)(brow[j] + k0 + 16));
      acc[j] = __builtin_amdgcn_wmma_f32_16x16x32_f16(false, af, false, bf,
                                                      (short)0, acc[j], false, false);
    }
  }

  float* cbase = C + (size_t)(m0 + (hi ? 8 : 0)) * M + (lane & 15);
  #pragma unroll
  for (int j = 0; j < NB; ++j) {
    float* crow = cbase + n0 + j * 16;
    #pragma unroll
    for (int v = 0; v < 8; ++v) crow[(size_t)v * M] = acc[j][v];
  }
}

// ---------------------------------------------------------------------------
__global__ void fill_f32(float* __restrict__ p, float v, int n) {
  int t = blockIdx.x * blockDim.x + threadIdx.x;
  if (t < n) p[t] = v;
}

__global__ void pad_x_f16(const float* __restrict__ x, _Float16* __restrict__ xh, int total) {
  int t = blockIdx.x * blockDim.x + threadIdx.x;
  if (t >= total) return;
  int i = t >> 5, c = t & 31;                 // pad 14 -> 32
  xh[t] = (c < 14) ? (_Float16)x[i * 14 + c] : (_Float16)0.f;
}

// W[K,M] f32 -> Wt[M,Kpad] f16 (transposed, K zero-padded to Kpad)
__global__ void convert_wt_pad(const float* __restrict__ W, _Float16* __restrict__ Wt,
                               int K, int Kpad, int M, int total) {
  int t = blockIdx.x * blockDim.x + threadIdx.x;
  if (t >= total) return;
  int m = t / Kpad, k = t % Kpad;
  Wt[t] = (k < K) ? (_Float16)W[(size_t)k * M + m] : (_Float16)0.f;
}

// per-column sum / sumsq (training-mode BN batch statistics)
__global__ void colstats(const float* __restrict__ X, float* __restrict__ stat,
                         int nRows, int C, int rpb) {
  int c = threadIdx.x;                         // blockDim.x == C
  int r0 = blockIdx.x * rpb;
  int r1 = min(r0 + rpb, nRows);
  float s = 0.f, ss = 0.f;
  for (int r = r0; r < r1; ++r) {
    float v = X[(size_t)r * C + c];
    s += v; ss += v * v;
  }
  atomicAdd(&stat[c], s);
  atomicAdd(&stat[C + c], ss);
}

// y = g*(x-mean)*rsqrt(var+eps)+b [+relu]; optional f16 copy for next WMMA GEMM
__global__ void bn_apply(const float* __restrict__ X, const float* __restrict__ stat,
                         const float* __restrict__ g, const float* __restrict__ b,
                         float* __restrict__ Y, _Float16* __restrict__ Yh,
                         int total, int C, float invn, int relu) {
  int t = blockIdx.x * blockDim.x + threadIdx.x;
  if (t >= total) return;
  int c = t % C;
  float mean = stat[c] * invn;
  float var  = stat[C + c] * invn - mean * mean;
  float y = g[c] * (X[t] - mean) * rsqrtf(var + EPSBN) + b[c];
  if (relu) y = fmaxf(y, 0.f);
  Y[t] = y;
  if (Yh) Yh[t] = (_Float16)y;
}

// e_src/e_dst: per (node, head) dot over C channels
__global__ void attn_coef(const float* __restrict__ hh, const float* __restrict__ as,
                          const float* __restrict__ ad, float* __restrict__ es,
                          float* __restrict__ ed, int total, int C) {
  int t = blockIdx.x * blockDim.x + threadIdx.x;
  if (t >= total) return;
  int n = t / HEADS, h = t % HEADS;
  const float* row = hh + (size_t)n * HEADS * C + h * C;
  float s1 = 0.f, s2 = 0.f;
  for (int c = 0; c < C; ++c) {
    float v = row[c];
    s1 += v * as[h * C + c];
    s2 += v * ad[h * C + c];
  }
  es[t] = s1; ed[t] = s2;
}

__device__ __forceinline__ void edge_sd(const int* __restrict__ ei, int i, int E,
                                        int& s, int& d) {
  if (i < E) { s = ei[i]; d = ei[E + i]; }
  else       { s = d = i - E; }              // PyG-style self loops
}

__global__ void edge_max(const int* __restrict__ ei, const float* __restrict__ es,
                         const float* __restrict__ ed, float* __restrict__ smax,
                         int EN, int E) {
  int t = blockIdx.x * blockDim.x + threadIdx.x;
  if (t >= EN * HEADS) return;
  int i = t / HEADS, h = t % HEADS, s, d;
  edge_sd(ei, i, E, s, d);
  float e = leakyf(es[s * HEADS + h] + ed[d * HEADS + h]);
  atomicMaxF(&smax[d * HEADS + h], e);
}

__global__ void edge_sum(const int* __restrict__ ei, const float* __restrict__ es,
                         const float* __restrict__ ed, const float* __restrict__ smax,
                         float* __restrict__ ssum, int EN, int E) {
  int t = blockIdx.x * blockDim.x + threadIdx.x;
  if (t >= EN * HEADS) return;
  int i = t / HEADS, h = t % HEADS, s, d;
  edge_sd(ei, i, E, s, d);
  float e = leakyf(es[s * HEADS + h] + ed[d * HEADS + h]);
  atomicAdd(&ssum[d * HEADS + h], __expf(e - smax[d * HEADS + h]));
}

__global__ void edge_agg(const int* __restrict__ ei, const float* __restrict__ es,
                         const float* __restrict__ ed, const float* __restrict__ smax,
                         const float* __restrict__ ssum, const float* __restrict__ hh,
                         float* __restrict__ hout, int EN, int E, int C) {
  int t = blockIdx.x * blockDim.x + threadIdx.x;
  if (t >= EN * HEADS) return;
  int i = t / HEADS, h = t % HEADS, s, d;
  edge_sd(ei, i, E, s, d);
  int dh = d * HEADS + h;
  float e = leakyf(es[s * HEADS + h] + ed[dh]);
  float alpha = __expf(e - smax[dh]) / ssum[dh];
  const float* srow = hh   + (size_t)s * HEADS * C + h * C;
  float*       drow = hout + (size_t)d * HEADS * C + h * C;
  __builtin_prefetch(srow, 0, 1);
  for (int c = 0; c < C; ++c) atomicAdd(&drow[c], srow[c] * alpha);
}

// gate scalar: out[i] = X[i,:] . w + b
__global__ void row_dot(const float* __restrict__ X, const float* __restrict__ w,
                        const float* __restrict__ b, float* __restrict__ out,
                        int n, int K) {
  int i = blockIdx.x * blockDim.x + threadIdx.x;
  if (i >= n) return;
  float s = b ? b[0] : 0.f;
  for (int k = 0; k < K; ++k) s += X[(size_t)i * K + k] * w[k];
  out[i] = s;
}

__global__ void graph_max(const float* __restrict__ gate, const int* __restrict__ bid,
                          float* __restrict__ gmax, int n) {
  int i = blockIdx.x * blockDim.x + threadIdx.x;
  if (i < n) atomicMaxF(&gmax[bid[i]], gate[i]);
}

__global__ void graph_sum(const float* __restrict__ gate, const int* __restrict__ bid,
                          const float* __restrict__ gmax, float* __restrict__ gsum, int n) {
  int i = blockIdx.x * blockDim.x + threadIdx.x;
  if (i < n) atomicAdd(&gsum[bid[i]], __expf(gate[i] - gmax[bid[i]]));
}

__global__ void graph_pool(const float* __restrict__ h, const float* __restrict__ gate,
                           const int* __restrict__ bid, const float* __restrict__ gmax,
                           const float* __restrict__ gsum, float* __restrict__ pooled,
                           int n, int C) {
  int i = blockIdx.x * blockDim.x + threadIdx.x;
  if (i >= n) return;
  int g = bid[i];
  float a = __expf(gate[i] - gmax[g]) / gsum[g];
  const float* row = h + (size_t)i * C;
  float* prow = pooled + (size_t)g * C;
  for (int c = 0; c < C; ++c) atomicAdd(&prow[c], row[c] * a);
}

// small dense layer (classifier tail, G=1024 rows): Y = X @ W (+b)
__global__ void dense(const float* __restrict__ X, const float* __restrict__ W,
                      const float* __restrict__ b, float* __restrict__ Y,
                      int n, int K, int M) {
  int t = blockIdx.x * blockDim.x + threadIdx.x;
  if (t >= n * M) return;
  int i = t / M, m = t % M;
  float s = b ? b[m] : 0.f;
  for (int k = 0; k < K; ++k) s += X[(size_t)i * K + k] * W[(size_t)k * M + m];
  Y[t] = s;
}

// ---------------------------------------------------------------------------
#define L1D(kern, total, ...) \
  kern<<<((total) + 255) / 256, 256, 0, stream>>>(__VA_ARGS__)

extern "C" void kernel_launch(void* const* d_in, const int* in_sizes, int n_in,
                              void* d_out, int out_size, void* d_ws, size_t ws_size,
                              hipStream_t stream) {
  const int N = N_NODES, E = N_EDGES, G = N_GRAPHS;
  const int EN = E + N;
  const float invN = 1.f / (float)N;
  const float invG = 1.f / (float)G;

  // ---- inputs (setup_inputs dict order, tuples flattened) ----
  const float* x     = (const float*)d_in[0];
  const int*   ei    = (const int*)d_in[1];
  const int*   bid   = (const int*)d_in[2];
  const float* pW    = (const float*)d_in[3];   // 14x256 (bias d_in[4] cancels in BN)
  const float* pG    = (const float*)d_in[5];
  const float* pB    = (const float*)d_in[6];
  const float* gatW[3]  = {(const float*)d_in[7],  (const float*)d_in[11], (const float*)d_in[15]};
  const float* gatAs[3] = {(const float*)d_in[8],  (const float*)d_in[12], (const float*)d_in[16]};
  const float* gatAd[3] = {(const float*)d_in[9],  (const float*)d_in[13], (const float*)d_in[17]};
  // bg (10,14,18) cancel in the following BN
  const float* bnG[3] = {(const float*)d_in[19], (const float*)d_in[21], (const float*)d_in[23]};
  const float* bnB[3] = {(const float*)d_in[20], (const float*)d_in[22], (const float*)d_in[24]};
  const float* gW1 = (const float*)d_in[25];    // gb1 (26) cancels in BN
  const float* gG  = (const float*)d_in[27];
  const float* gBt = (const float*)d_in[28];
  const float* gW2 = (const float*)d_in[29];
  const float* gb2 = (const float*)d_in[30];
  const float* cW1 = (const float*)d_in[31];    // cb1 (32) cancels
  const float* cG1 = (const float*)d_in[33];
  const float* cBt1= (const float*)d_in[34];
  const float* cW2 = (const float*)d_in[35];    // cb2 (36) cancels
  const float* cG2 = (const float*)d_in[37];
  const float* cBt2= (const float*)d_in[38];
  const float* cW3 = (const float*)d_in[39];
  const float* cb3 = (const float*)d_in[40];
  float* out = (float*)d_out;

  // ---- workspace carve (256B aligned slabs) ----
  char* wp = (char*)d_ws; size_t off = 0;
  auto carve = [&](size_t bytes) -> void* {
    void* p = wp + off; off = (off + bytes + 255) & ~(size_t)255; return p;
  };
  float*    h    = (float*)carve((size_t)N * 256 * 4);
  float*    hh   = (float*)carve((size_t)N * 256 * 4);
  _Float16* hA   = (_Float16*)carve((size_t)N * 256 * 2);
  _Float16* wB   = (_Float16*)carve((size_t)256 * 256 * 2);
  float*    stat = (float*)carve(512 * 4);
  float*    esrc = (float*)carve((size_t)N * HEADS * 4);
  float*    edst = (float*)carve((size_t)N * HEADS * 4);
  float*    smax = (float*)carve((size_t)N * HEADS * 4);
  float*    ssum = (float*)carve((size_t)N * HEADS * 4);
  float*    gate = (float*)carve((size_t)N * 4);
  float*    gmax = (float*)carve((size_t)G * 4);
  float*    gsum = (float*)carve((size_t)G * 4);
  float*    pool = (float*)carve((size_t)G * 64 * 4);
  float*    zg1  = (float*)carve((size_t)G * 32 * 4);
  float*    zg2  = (float*)carve((size_t)G * 16 * 4);

  // K is compile-time at every call site; NB=4 strip (16x64) except gate (M=32 -> NB=2)
  auto gemm = [&](const _Float16* A_, const _Float16* B_, float* C_,
                  int n, int K, int M) {
    if (M % 64 == 0) {
      int blocks = ((n / 16) * (M / 64) + 7) / 8;
      switch (K) {
        case 32:  wmma_gemm_f16<4, 32> <<<blocks, 256, 0, stream>>>(A_, B_, C_, n, M); break;
        case 128: wmma_gemm_f16<4, 128><<<blocks, 256, 0, stream>>>(A_, B_, C_, n, M); break;
        default:  wmma_gemm_f16<4, 256><<<blocks, 256, 0, stream>>>(A_, B_, C_, n, M); break;
      }
    } else {
      int blocks = ((n / 16) * (M / 32) + 7) / 8;
      wmma_gemm_f16<2, 64><<<blocks, 256, 0, stream>>>(A_, B_, C_, n, M);  // gate: K=64
    }
  };

  // ================= projection: h = relu(bn(x @ pW)) =================
  L1D(pad_x_f16, N * 32, x, hA, N * 32);
  L1D(convert_wt_pad, 256 * 32, pW, wB, 14, 32, 256, 256 * 32);
  gemm(hA, wB, hh, N, 32, 256);
  hipMemsetAsync(stat, 0, 2 * 256 * 4, stream);
  colstats<<<(N + 127) / 128, 256, 0, stream>>>(hh, stat, N, 256, 128);
  L1D(bn_apply, N * 256, hh, stat, pG, pB, h, hA, N * 256, 256, invN, 1);

  // ================= 3 GAT layers =================
  const int Kin[3]  = {256, 256, 128};
  const int Mout[3] = {256, 128, 64};
  const int Cph[3]  = {32, 16, 8};
  for (int l = 0; l < 3; ++l) {
    const int K = Kin[l], M = Mout[l], C = Cph[l];
    L1D(convert_wt_pad, M * K, gatW[l], wB, K, K, M, M * K);
    gemm(hA, wB, hh, N, K, M);
    L1D(attn_coef, N * HEADS, hh, gatAs[l], gatAd[l], esrc, edst, N * HEADS, C);
    L1D(fill_f32, N * HEADS, smax, -INFINITY, N * HEADS);
    hipMemsetAsync(ssum, 0, (size_t)N * HEADS * 4, stream);
    L1D(edge_max, EN * HEADS, ei, esrc, edst, smax, EN, E);
    L1D(edge_sum, EN * HEADS, ei, esrc, edst, smax, ssum, EN, E);
    hipMemsetAsync(h, 0, (size_t)N * M * 4, stream);
    L1D(edge_agg, EN * HEADS, ei, esrc, edst, smax, ssum, hh, h, EN, E, C);
    hipMemsetAsync(stat, 0, 2 * M * 4, stream);
    colstats<<<(N + 127) / 128, M, 0, stream>>>(h, stat, N, M, 128);
    L1D(bn_apply, N * M, h, stat, bnG[l], bnB[l], h, hA, N * M, M, invN, 1);
  }

  // ================= gate + attention pooling =================
  L1D(convert_wt_pad, 32 * 64, gW1, wB, 64, 64, 32, 32 * 64);
  gemm(hA, wB, hh, N, 64, 32);
  hipMemsetAsync(stat, 0, 2 * 32 * 4, stream);
  colstats<<<(N + 127) / 128, 32, 0, stream>>>(hh, stat, N, 32, 128);
  L1D(bn_apply, N * 32, hh, stat, gG, gBt, hh, (_Float16*)nullptr, N * 32, 32, invN, 1);
  L1D(row_dot, N, hh, gW2, gb2, gate, N, 32);
  L1D(fill_f32, G, gmax, -INFINITY, G);
  hipMemsetAsync(gsum, 0, (size_t)G * 4, stream);
  hipMemsetAsync(pool, 0, (size_t)G * 64 * 4, stream);
  L1D(graph_max, N, gate, bid, gmax, N);
  L1D(graph_sum, N, gate, bid, gmax, gsum, N);
  L1D(graph_pool, N, h, gate, bid, gmax, gsum, pool, N, 64);

  // ================= classifier (G=1024 rows, scalar kernels) =================
  L1D(dense, G * 32, pool, cW1, (const float*)nullptr, zg1, G, 64, 32);
  hipMemsetAsync(stat, 0, 2 * 32 * 4, stream);
  colstats<<<(G + 127) / 128, 32, 0, stream>>>(zg1, stat, G, 32, 128);
  L1D(bn_apply, G * 32, zg1, stat, cG1, cBt1, zg1, (_Float16*)nullptr, G * 32, 32, invG, 1);
  L1D(dense, G * 16, zg1, cW2, (const float*)nullptr, zg2, G, 32, 16);
  hipMemsetAsync(stat, 0, 2 * 16 * 4, stream);
  colstats<<<(G + 127) / 128, 16, 0, stream>>>(zg2, stat, G, 16, 128);
  L1D(bn_apply, G * 16, zg2, stat, cG2, cBt2, zg2, (_Float16*)nullptr, G * 16, 16, invG, 1);
  L1D(dense, G * 2, zg2, cW3, cb3, out, G, 16, 2);
}
